// LinearAttention_10144712753263
// MI455X (gfx1250) — matrix operand order repaired and checked by reading, not compiled, not run on hardware
//
#include <hip/hip_runtime.h>

// ---------------------------------------------------------------------------
// Linear attention (softmax feature map) for MI455X / gfx1250.
//   B=8, L=4096, H=16, E=64, fp32 in/out.
//   Phase 1: KV[e][d] = sum_l softmax(k)[l][e] * v[l][d]   (per b,h)  + k_sum
//   Phase 2: out[l][d] = z_l * sum_e softmax(q)[l][e] * KV[e][d]
// Matrix contractions run on v_wmma_f32_16x16x32_f16 (f16 operands, f32 acc).
// ---------------------------------------------------------------------------

typedef __attribute__((ext_vector_type(16))) _Float16 v16h;
typedef __attribute__((ext_vector_type(8)))  float    v8f;

#define B_      8
#define L_      4096
#define H_      16
#define E_      64
#define CHUNK   128            // L rows per block
#define NCHUNK  (L_ / CHUNK)   // 32
#define LPAD    8              // LDS padding (halves) against bank conflicts
#define EPS_    1e-6f
#define NEGC_   -20.0f

__device__ __forceinline__ float softplus_f(float x) {
    return (x > 20.0f) ? x : log1pf(expf(x));
}

__device__ __forceinline__ float wave_max(float v) {
#pragma unroll
    for (int m = 16; m >= 1; m >>= 1) v = fmaxf(v, __shfl_xor(v, m, 32));
    return v;
}
__device__ __forceinline__ float wave_sum(float v) {
#pragma unroll
    for (int m = 16; m >= 1; m >>= 1) v += __shfl_xor(v, m, 32);
    return v;
}

// ---------------------------------------------------------------------------
__global__ void zero_ws_kernel(float* __restrict__ p, int n) {
    int i = blockIdx.x * blockDim.x + threadIdx.x;
    if (i < n) p[i] = 0.0f;
}

// ---------------------------------------------------------------------------
// Phase 1: partial KV (64x64) and k_sum (64) per (b,h), reduced via f32
// global atomics into workspace (resident in L2: ~2 MB total).
// ---------------------------------------------------------------------------
__global__ __launch_bounds__(256) void la_phase1(
        const float* __restrict__ keys,
        const float* __restrict__ values,
        const float* __restrict__ delta1,
        float* __restrict__ kv_ws,      // [B*H][64][64]
        float* __restrict__ ksum_ws)    // [B*H][64]
{
    __shared__ _Float16 kT[E_][CHUNK + LPAD];   // kT[e][l_local]
    __shared__ _Float16 vT[E_][CHUNK + LPAD];   // vT[d][l_local]
    __shared__ float    ksum_l[E_];

    const int chunk = blockIdx.x;
    const int h     = blockIdx.y;
    const int b     = blockIdx.z;
    const int tid   = threadIdx.x;
    const int lane  = tid & 31;
    const int wave  = tid >> 5;

    const float itemp = 1.0f / softplus_f(delta1[0]);

    if (tid < E_) ksum_l[tid] = 0.0f;
    __syncthreads();

    // ---- softmax(k) staged transposed into LDS; v staged transposed -------
    const int rows_per_wave = CHUNK / 8;   // 16
#pragma unroll 1
    for (int r = 0; r < rows_per_wave; ++r) {
        const int lrow = wave * rows_per_wave + r;
        const int l    = chunk * CHUNK + lrow;
        const size_t base = ((size_t)(b * L_ + l) * H_ + h) * E_;

        float k0 = keys[base + lane];
        float k1 = keys[base + lane + 32];
        k0 = (k0 < 0.0f ? NEGC_ : k0) * itemp;
        k1 = (k1 < 0.0f ? NEGC_ : k1) * itemp;
        const float mx = wave_max(fmaxf(k0, k1));
        const float e0 = expf(k0 - mx);
        const float e1 = expf(k1 - mx);
        const float inv = 1.0f / wave_sum(e0 + e1);
        const float sk0 = e0 * inv, sk1 = e1 * inv;
        kT[lane     ][lrow] = (_Float16)sk0;
        kT[lane + 32][lrow] = (_Float16)sk1;
        atomicAdd(&ksum_l[lane],      sk0);
        atomicAdd(&ksum_l[lane + 32], sk1);

        vT[lane     ][lrow] = (_Float16)values[base + lane];
        vT[lane + 32][lrow] = (_Float16)values[base + lane + 32];
    }
    __syncthreads();

    // ---- WMMA: KV[e][d] += kT[e][:] . vT[d][:] ----------------------------
    const int bh = b * H_ + h;
    float* kv_base = kv_ws + (size_t)bh * E_ * E_;
    const int koffA = (lane < 16) ? 0 : 8;    // A-fragment K offset (16b A 16x32)
    const int koffB = (lane < 16) ? 0 : 8;    // B-fragment K offset (16b B 32x16)

#pragma unroll
    for (int t = 0; t < 2; ++t) {
        const int tile = wave * 2 + t;        // 0..15 -> 4x4 tile grid
        const int me = tile >> 2, nd = tile & 3;
        const int erow = me * 16 + (lane & 15);
        const int ncol = nd * 16 + (lane & 15);
        v8f acc = {};
#pragma unroll
        for (int l0 = 0; l0 < CHUNK; l0 += 32) {
            v16h a, bf;
#pragma unroll
            for (int j = 0; j < 8; ++j) {
                const int ks = ((j < 4) ? 2 * j : 16 + 2 * (j - 4)) + koffA;
                a[2 * j]     = kT[erow][l0 + ks];
                a[2 * j + 1] = kT[erow][l0 + ks + 1];
            }
#pragma unroll
            for (int i = 0; i < 16; ++i)
                bf[i] = vT[ncol][l0 + koffB + ((i < 8) ? i : 8 + i)];
            acc = __builtin_amdgcn_wmma_f32_16x16x32_f16(
                      false, a, false, bf, (short)0, acc, false, false);
        }
        const int mbase = me * 16 + ((lane < 16) ? 0 : 8);
#pragma unroll
        for (int r = 0; r < 8; ++r)
            atomicAdd(&kv_base[(mbase + r) * E_ + ncol], acc[r]);
    }

    __syncthreads();
    if (tid < E_) atomicAdd(&ksum_ws[bh * E_ + tid], ksum_l[tid]);
}

// ---------------------------------------------------------------------------
// Phase 2: out[l][d] = z_l * (softmax(q)[l][:] @ KV[:][d])
// ---------------------------------------------------------------------------
__global__ __launch_bounds__(256) void la_phase2(
        const float* __restrict__ queries,
        const float* __restrict__ delta1,
        const float* __restrict__ kv_ws,
        const float* __restrict__ ksum_ws,
        float* __restrict__ out)
{
    __shared__ _Float16 qh[CHUNK][E_ + LPAD];   // qh[l_local][e] (row major)
    __shared__ _Float16 kvT[E_][E_ + LPAD];     // kvT[d][e]  (B operand, contiguous K)
    __shared__ float    zl[CHUNK];
    __shared__ float    ksmem[E_];

    const int chunk = blockIdx.x;
    const int h     = blockIdx.y;
    const int b     = blockIdx.z;
    const int tid   = threadIdx.x;
    const int lane  = tid & 31;
    const int wave  = tid >> 5;
    const int bh    = b * H_ + h;

    const float itemp = 1.0f / softplus_f(delta1[0]);

    // ---- load KV (f32 -> f16, transposed) and k_sum -----------------------
    const float* kvg = kv_ws + (size_t)bh * E_ * E_;
#pragma unroll
    for (int i = 0; i < (E_ * E_) / 256; ++i) {
        const int idx = tid + 256 * i;
        const int e = idx >> 6, d = idx & 63;
        kvT[d][e] = (_Float16)kvg[idx];
    }
    if (tid < E_) ksmem[tid] = ksum_ws[bh * E_ + tid];
    __syncthreads();

    // ---- softmax(q) + z ---------------------------------------------------
    const int rows_per_wave = CHUNK / 8;   // 16
#pragma unroll 1
    for (int r = 0; r < rows_per_wave; ++r) {
        const int lrow = wave * rows_per_wave + r;
        const int l    = chunk * CHUNK + lrow;
        const size_t base = ((size_t)(b * L_ + l) * H_ + h) * E_;

        float q0 = queries[base + lane];
        float q1 = queries[base + lane + 32];
        q0 = (q0 < 0.0f ? NEGC_ : q0) * itemp;
        q1 = (q1 < 0.0f ? NEGC_ : q1) * itemp;
        const float mx = wave_max(fmaxf(q0, q1));
        const float e0 = expf(q0 - mx);
        const float e1 = expf(q1 - mx);
        const float inv = 1.0f / wave_sum(e0 + e1);
        const float sq0 = e0 * inv, sq1 = e1 * inv;
        qh[lrow][lane]      = (_Float16)sq0;
        qh[lrow][lane + 32] = (_Float16)sq1;

        const float dot = wave_sum(sq0 * ksmem[lane] + sq1 * ksmem[lane + 32]);
        if (lane == 0) zl[lrow] = 1.0f / (dot + EPS_);
    }
    __syncthreads();

    // ---- WMMA: OUT = Q @ KV, scaled by z ----------------------------------
    const int koffA = (lane < 16) ? 0 : 8;
    const int koffB = (lane < 16) ? 0 : 8;

#pragma unroll
    for (int t = 0; t < 4; ++t) {
        const int tile  = wave + 8 * t;       // 0..31 -> 8(l) x 4(d) tile grid
        const int ltile = tile >> 2, dtile = tile & 3;
        const int mrow = ltile * 16 + (lane & 15);
        const int ncol = dtile * 16 + (lane & 15);
        v8f acc = {};
#pragma unroll
        for (int e0i = 0; e0i < E_; e0i += 32) {
            v16h a, bf;
#pragma unroll
            for (int j = 0; j < 8; ++j) {
                const int ks = ((j < 4) ? 2 * j : 16 + 2 * (j - 4)) + koffA;
                a[2 * j]     = qh[mrow][e0i + ks];
                a[2 * j + 1] = qh[mrow][e0i + ks + 1];
            }
#pragma unroll
            for (int i = 0; i < 16; ++i)
                bf[i] = kvT[ncol][e0i + koffB + ((i < 8) ? i : 8 + i)];
            acc = __builtin_amdgcn_wmma_f32_16x16x32_f16(
                      false, a, false, bf, (short)0, acc, false, false);
        }
        const int mbase = ltile * 16 + ((lane < 16) ? 0 : 8);
#pragma unroll
        for (int r = 0; r < 8; ++r) {
            const int lr = mbase + r;
            const int l  = chunk * CHUNK + lr;
            out[((size_t)(b * L_ + l) * H_ + h) * E_ + ncol] = acc[r] * zl[lr];
        }
    }
}

// ---------------------------------------------------------------------------
extern "C" void kernel_launch(void* const* d_in, const int* in_sizes, int n_in,
                              void* d_out, int out_size, void* d_ws, size_t ws_size,
                              hipStream_t stream) {
    const float* queries = (const float*)d_in[0];
    const float* keys    = (const float*)d_in[1];
    const float* values  = (const float*)d_in[2];
    const float* delta1  = (const float*)d_in[3];
    float* out = (float*)d_out;

    float* kv_ws   = (float*)d_ws;                       // B*H*64*64 floats
    float* ksum_ws = kv_ws + (size_t)B_ * H_ * E_ * E_;  // B*H*64 floats
    const int ws_floats = B_ * H_ * E_ * E_ + B_ * H_ * E_;

    zero_ws_kernel<<<(ws_floats + 255) / 256, 256, 0, stream>>>(kv_ws, ws_floats);

    dim3 grid(NCHUNK, H_, B_);
    la_phase1<<<grid, 256, 0, stream>>>(keys, values, delta1, kv_ws, ksum_ws);
    la_phase2<<<grid, 256, 0, stream>>>(queries, delta1, kv_ws, ksum_ws, out);
    (void)in_sizes; (void)n_in; (void)out_size; (void)ws_size;
}
